// MPLayer_NE_49701361549769
// MI455X (gfx1250) — compile-verified, matrix operand order checked by reading, not run on hardware
//
#include <hip/hip_runtime.h>

// MI455X / gfx1250: wave32, WMMA 16x16x32 bf16 (f32 accumulate).
// Layered-DAG propagation: 15 sequential levels, each = GEMM(+bias+relu) ->
// gather/mean over 16 parents -> GEMM(+bias+relu+residual).
// Compute (64 GFLOP total) is trivial vs WMMA peak; all intermediates
// (msg/z/y_bf at 8 MB each) are L2 resident (192 MB L2). The GEMM k-loop is
// software-pipelined (double-buffered fragments) so global loads for step
// ks+1 stay in flight across the 4 WMMAs of step ks.

typedef __attribute__((ext_vector_type(16))) __bf16 v16bf;
typedef __attribute__((ext_vector_type(8)))  __bf16 v8bf;
typedef __attribute__((ext_vector_type(8)))  float  v8f;

#define NODES 16384
#define DIM   256
#define KIN   16
#define LVL   16

// ---- A-fragment loader: 16x32 bf16 A layout (ISA 7.12.2):
// lane(0-15)=row M, khalf=lane>>4; per lane: K = kb..kb+7 (regs 0-3) and
// K = kb+16..kb+23 (regs 4-7), kb = kstep*32 + khalf*8. Both runs contiguous,
// so an A fragment is two 16-byte loads, zero VALU.
static __device__ inline v16bf frag_from(const __bf16* p) {
  v8bf lo = *(const v8bf*)p;
  v8bf hi = *(const v8bf*)(p + 16);
  v16bf a;
#pragma unroll
  for (int i = 0; i < 8; ++i) { a[i] = lo[i]; a[i + 8] = hi[i]; }
  return a;
}

// ---- One-shot repack of W (f32 row-major, [K=256][N=256]) into per-lane
// B-fragment order so the GEMM's B load is a single contiguous 32B load.
// Fragment f = nTile*8 + kstep; within: 32 lanes x 16 bf16.
// B (32x16, KxN) layout mirror of A: lane(0-15)=col N, elems 0..7 -> K=kb..kb+7,
// elems 8..15 -> K=kb+16..kb+23, kb = khalf*8.
__global__ void convert_weights_kernel(const float* __restrict__ W1,
                                       const float* __restrict__ W2,
                                       __bf16* __restrict__ W1f,
                                       __bf16* __restrict__ W2f) {
  const int tid = blockIdx.x * blockDim.x + threadIdx.x;   // 0..65535
  const int f   = tid >> 9;          // fragment id (512 elems each)
  const int rem = tid & 511;
  const int ln  = rem >> 4;          // lane 0..31
  const int e   = rem & 15;          // element 0..15
  const int nTile = f >> 3;
  const int ks    = f & 7;
  const int col   = nTile * 16 + (ln & 15);
  const int kk    = (e < 8) ? ((ln >> 4) * 8 + e)
                            : (16 + (ln >> 4) * 8 + (e - 8));
  const int src   = (ks * 32 + kk) * DIM + col;
  W1f[tid] = (__bf16)W1[src];
  W2f[tid] = (__bf16)W2[src];
}

// ---- Seed y_bf = bf16(x[0]) for level 0. 8 elems/thread, vector I/O.
__global__ void init_ybf_kernel(const float* __restrict__ x0,
                                __bf16* __restrict__ ybf) {
  const size_t i = ((size_t)blockIdx.x * blockDim.x + threadIdx.x) * 8;
  const v8f v = *(const v8f*)(x0 + i);
  v8bf o;
#pragma unroll
  for (int j = 0; j < 8; ++j) o[j] = (__bf16)v[j];
  *(v8bf*)(ybf + i) = o;
}

// ---- Fused GEMM: relu(A[16384x256]_bf16 * W[256x256]_bf16 + bias) (+resid)
// One wave owns a 64x16 output slab: 4 f32 accumulators, K-loop of 8, with
// next-step fragments double-buffered so loads overlap the WMMAs.
// 4096 waves total = 512 blocks x 8 waves. EXEC all-ones (WMMA requirement).
__global__ void gemm_bias_relu_kernel(const __bf16* __restrict__ A,
                                      const __bf16* __restrict__ Wf,
                                      const float* __restrict__ bias,
                                      const float* __restrict__ resid,  // nullable
                                      float* __restrict__ outF,         // nullable
                                      __bf16* __restrict__ outB) {      // nullable
  const int lane  = threadIdx.x & 31;
  const int wave  = (blockIdx.x * blockDim.x + threadIdx.x) >> 5;  // 0..4095
  const int mg    = wave >> 4;        // 0..255  (64-row group)
  const int nt    = wave & 15;        // 0..15   (16-col tile)
  const int rowB  = mg * 64;
  const int arow  = lane & 15;
  const int khalf = lane >> 4;

  // Per-lane base pointers (k offset added per step).
  const __bf16* Ar0 = A + (size_t)(rowB +  0 + arow) * DIM + khalf * 8;
  const __bf16* Ar1 = A + (size_t)(rowB + 16 + arow) * DIM + khalf * 8;
  const __bf16* Ar2 = A + (size_t)(rowB + 32 + arow) * DIM + khalf * 8;
  const __bf16* Ar3 = A + (size_t)(rowB + 48 + arow) * DIM + khalf * 8;
  const __bf16* Bp  = Wf + (size_t)(nt << 3) * 512 + lane * 16;  // frag stride 512

  v8f acc[4] = {v8f{}, v8f{}, v8f{}, v8f{}};

  // Prologue: fragments for k-step 0.
  v16bf b  = *(const v16bf*)Bp;
  v16bf a0 = frag_from(Ar0);
  v16bf a1 = frag_from(Ar1);
  v16bf a2 = frag_from(Ar2);
  v16bf a3 = frag_from(Ar3);

#pragma unroll
  for (int ks = 0; ks < 8; ++ks) {
    v16bf bn = b, n0 = a0, n1 = a1, n2 = a2, n3 = a3;
    if (ks < 7) {
      const int kn = (ks + 1) * 32;
      bn = *(const v16bf*)(Bp + (size_t)(ks + 1) * 512);
      n0 = frag_from(Ar0 + kn);
      n1 = frag_from(Ar1 + kn);
      n2 = frag_from(Ar2 + kn);
      n3 = frag_from(Ar3 + kn);
    }
    acc[0] = __builtin_amdgcn_wmma_f32_16x16x32_bf16(false, a0, false, b, (short)0, acc[0], false, false);
    acc[1] = __builtin_amdgcn_wmma_f32_16x16x32_bf16(false, a1, false, b, (short)0, acc[1], false, false);
    acc[2] = __builtin_amdgcn_wmma_f32_16x16x32_bf16(false, a2, false, b, (short)0, acc[2], false, false);
    acc[3] = __builtin_amdgcn_wmma_f32_16x16x32_bf16(false, a3, false, b, (short)0, acc[3], false, false);
    b = bn; a0 = n0; a1 = n1; a2 = n2; a3 = n3;
  }

  // C/D layout (ISA 7.12.2): lane(0-15)=col n, reg r -> row khalf*8 + r.
  const int n  = nt * 16 + (lane & 15);
  const float bv = bias[n];
#pragma unroll
  for (int t = 0; t < 4; ++t) {
    const int rbase = rowB + t * 16 + khalf * 8;
#pragma unroll
    for (int r = 0; r < 8; ++r) {
      const size_t o = (size_t)(rbase + r) * DIM + n;
      float v = acc[t][r] + bv;
      v = v > 0.0f ? v : 0.0f;
      if (resid) v += resid[o];
      if (outF)  outF[o] = v;
      if (outB)  outB[o] = (__bf16)v;
    }
  }
}

// ---- z[i,:] = mean_k msg[idx[i,k],:]  (bf16 in, f32 accumulate, bf16 out)
// One wave per node; each lane owns 8 contiguous feature channels (16B loads).
// msg is 8 MB -> fully L2 resident; prefetch the next parent row while
// accumulating the current one (global_prefetch_b8).
__global__ void gather_mean_kernel(const __bf16* __restrict__ msg,
                                   const int* __restrict__ idx,
                                   __bf16* __restrict__ z) {
  const int lane  = threadIdx.x & 31;
  const int node  = (blockIdx.x * blockDim.x + threadIdx.x) >> 5;  // 0..16383
  const int* ip   = idx + (size_t)node * KIN;
  const int dbase = lane * 8;
  float s[8] = {0.f, 0.f, 0.f, 0.f, 0.f, 0.f, 0.f, 0.f};
  int src = ip[0];
#pragma unroll
  for (int k = 0; k < KIN; ++k) {
    const v8bf m = *(const v8bf*)(msg + (size_t)src * DIM + dbase);
    if (k + 1 < KIN) {
      src = ip[k + 1];
      __builtin_prefetch(msg + (size_t)src * DIM + dbase, 0, 3);
    }
#pragma unroll
    for (int j = 0; j < 8; ++j) s[j] += (float)m[j];
  }
  v8bf zo;
#pragma unroll
  for (int j = 0; j < 8; ++j) zo[j] = (__bf16)(s[j] * (1.0f / KIN));
  *(v8bf*)(z + (size_t)node * DIM + dbase) = zo;
}

extern "C" void kernel_launch(void* const* d_in, const int* in_sizes, int n_in,
                              void* d_out, int out_size, void* d_ws, size_t ws_size,
                              hipStream_t stream) {
  const float* x   = (const float*)d_in[0];   // [L,N,D]
  const int*   idx = (const int*)d_in[1];     // [L-1,N,K]
  const float* W1  = (const float*)d_in[2];   // [D,D]
  const float* b1  = (const float*)d_in[3];   // [D]
  const float* W2  = (const float*)d_in[4];   // [D,D]
  const float* b2  = (const float*)d_in[5];   // [D]
  float* out = (float*)d_out;                 // [L,N,D]

  const size_t ND2 = (size_t)NODES * DIM * 2; // bytes of one bf16 [N,D] plane
  char* ws = (char*)d_ws;
  __bf16* W1f = (__bf16*)ws;                              // 128 KB
  __bf16* W2f = (__bf16*)(ws + (size_t)131072);           // 128 KB
  __bf16* msg = (__bf16*)(ws + (size_t)262144);           // 8 MB
  __bf16* zb  = (__bf16*)(ws + (size_t)262144 + ND2);     // 8 MB
  __bf16* ybf = (__bf16*)(ws + (size_t)262144 + 2 * ND2); // 8 MB bf16 mirror of y

  // One-time weight repack into WMMA B-fragment order; seed level-0 state.
  convert_weights_kernel<<<256, 256, 0, stream>>>(W1, W2, W1f, W2f);
  init_ybf_kernel<<<2048, 256, 0, stream>>>(x, ybf);

  // Level 0: y0 = x[0] (no in-edges).
  hipMemcpyAsync(out, x, (size_t)NODES * DIM * sizeof(float),
                 hipMemcpyDeviceToDevice, stream);

  // Sequential topological levels; stream order provides the dependency.
  for (int l = 1; l < LVL; ++l) {
    // msg = relu(y_prev @ W1 + b1)   (A is the bf16 mirror of y_{l-1})
    gemm_bias_relu_kernel<<<512, 256, 0, stream>>>(
        ybf, W1f, b1, nullptr, nullptr, msg);
    // z = mean over parents
    gather_mean_kernel<<<2048, 256, 0, stream>>>(
        msg, idx + (size_t)(l - 1) * NODES * KIN, zb);
    // y = relu(z @ W2 + b2) + x[l]; write f32 output plane + bf16 mirror
    gemm_bias_relu_kernel<<<512, 256, 0, stream>>>(
        zb, W2f, b2,
        x + (size_t)l * NODES * DIM,
        out + (size_t)l * NODES * DIM,
        ybf);
  }
}
